// SliceFineLiMELinear_17325898072234
// MI455X (gfx1250) — matrix operand order repaired and checked by reading, not compiled, not run on hardware
//
#include <hip/hip_runtime.h>
#include <hip/hip_bf16.h>

// Problem constants (from reference)
#define B_    4
#define T_    2048
#define DIN_  2048
#define H_    2048
#define E_    8
#define RANK_ 64
#define NA_   512                 // anchors per batch: T/NGRAM, rem==0
#define SCALING_ 0.25f            // ALPHA/RANK = 16/64

typedef __attribute__((ext_vector_type(16))) __bf16       v16bf;
typedef __attribute__((ext_vector_type(8)))  float        v8f;
typedef __attribute__((ext_vector_type(8)))  unsigned int v8u;

// Split a pair of f32 into packed bf16 hi (RTZ) and bf16 lo (RTZ of remainder).
__device__ __forceinline__ void split_pair(float f0, float f1,
                                           unsigned& hi, unsigned& lo) {
    unsigned u0 = __float_as_uint(f0), u1 = __float_as_uint(f1);
    hi = (u0 >> 16) | (u1 & 0xFFFF0000u);
    float r0 = f0 - __uint_as_float(u0 & 0xFFFF0000u);
    float r1 = f1 - __uint_as_float(u1 & 0xFFFF0000u);
    unsigned v0 = __float_as_uint(r0), v1 = __float_as_uint(r1);
    lo = (v0 >> 16) | (v1 & 0xFFFF0000u);
}

// ---------------------------------------------------------------------------
// Prep pass: f32 array -> packed bf16 hi / bf16 lo arrays (2B/elem each).
// ---------------------------------------------------------------------------
__global__ void split_kernel(const float* __restrict__ src,
                             unsigned* __restrict__ hi, unsigned* __restrict__ lo,
                             int n8) {
    int i = blockIdx.x * blockDim.x + threadIdx.x;
    if (i >= n8) return;
    const float4* s = (const float4*)src;
    float4 a = s[2 * i], b = s[2 * i + 1];
    uint4 h, l;
    split_pair(a.x, a.y, h.x, l.x);
    split_pair(a.z, a.w, h.y, l.y);
    split_pair(b.x, b.y, h.z, l.z);
    split_pair(b.z, b.w, h.w, l.w);
    ((uint4*)hi)[i] = h;
    ((uint4*)lo)[i] = l;
}

// ---------------------------------------------------------------------------
// GEMM: base[M,H] = x[M,DIN] @ W[H,DIN]^T + b, M = B*T.
// 128x128 block tile, BK=32, 8 wave32s. DOUBLE-BUFFERED LDS: iteration k
// issues global_load_async_to_lds_b128 for tile k+1, then s_wait_asynccnt 8
// (in-order async completion => current tile landed) so the next tile's HBM
// traffic flies under the 24 WMMAs of the current tile.
// ---------------------------------------------------------------------------
#define BM  128
#define BN  128
#define BK  32
#define LDU 20                 // LDS row stride in uints (80B, conflict-free)
#define TILEU (BM * LDU)       // uints per buffer (2560)
#define TILEB (TILEU * 4)      // bytes per buffer (10240)

__global__ __launch_bounds__(256)
void gemm_kernel(const unsigned* __restrict__ Xh, const unsigned* __restrict__ Xl,
                 const unsigned* __restrict__ Wh, const unsigned* __restrict__ Wl,
                 const float* __restrict__ bias, float* __restrict__ out) {
    __shared__ unsigned As_hi[2 * TILEU];
    __shared__ unsigned As_lo[2 * TILEU];
    __shared__ unsigned Bs_hi[2 * TILEU];
    __shared__ unsigned Bs_lo[2 * TILEU];

    const int tid   = threadIdx.x;
    const int lane  = tid & 31;
    const int wave  = tid >> 5;
    const int half  = lane >> 4;   // 0 or 1
    const int l16   = lane & 15;
    const int waveM = wave & 3;    // 0..3 -> +32 rows each
    const int waveN = wave >> 2;   // 0..1 -> +64 cols each

    const int blockM = blockIdx.y * BM;
    const int blockN = blockIdx.x * BN;

    // Per-thread staging slots: tile = 128 rows x 64B, 4 x 16B chunks/row;
    // each thread owns two chunks per matrix.
    int srow[2], sc[2];
    srow[0] = tid >> 2;          sc[0] = tid & 3;
    srow[1] = (tid + 256) >> 2;  sc[1] = (tid + 256) & 3;

    const unsigned* gXh[2]; const unsigned* gXl[2];
    const unsigned* gWh[2]; const unsigned* gWl[2];
    unsigned lAh[2], lAl[2], lBh[2], lBl[2];
#pragma unroll
    for (int i = 0; i < 2; ++i) {
        size_t eb  = ((size_t)(blockM + srow[i]) * DIN_) >> 1;  // uint index
        size_t ebw = ((size_t)(blockN + srow[i]) * DIN_) >> 1;
        gXh[i] = Xh + eb  + sc[i] * 4;
        gXl[i] = Xl + eb  + sc[i] * 4;
        gWh[i] = Wh + ebw + sc[i] * 4;
        gWl[i] = Wl + ebw + sc[i] * 4;
        lAh[i] = (unsigned)(size_t)(As_hi + srow[i] * LDU + sc[i] * 4);
        lAl[i] = (unsigned)(size_t)(As_lo + srow[i] * LDU + sc[i] * 4);
        lBh[i] = (unsigned)(size_t)(Bs_hi + srow[i] * LDU + sc[i] * 4);
        lBl[i] = (unsigned)(size_t)(Bs_lo + srow[i] * LDU + sc[i] * 4);
    }

    // Issue one tile's 8 async loads into LDS buffer at byte offset boff,
    // then advance the global pointers by one K-step (64 B = 16 uints).
    auto stage = [&](unsigned boff) {
#pragma unroll
        for (int i = 0; i < 2; ++i) {
            asm volatile("global_load_async_to_lds_b128 %0, %1, off"
                         :: "v"(lAh[i] + boff), "v"(gXh[i]) : "memory");
            asm volatile("global_load_async_to_lds_b128 %0, %1, off"
                         :: "v"(lAl[i] + boff), "v"(gXl[i]) : "memory");
            asm volatile("global_load_async_to_lds_b128 %0, %1, off"
                         :: "v"(lBh[i] + boff), "v"(gWh[i]) : "memory");
            asm volatile("global_load_async_to_lds_b128 %0, %1, off"
                         :: "v"(lBl[i] + boff), "v"(gWl[i]) : "memory");
            gXh[i] += 16; gXl[i] += 16; gWh[i] += 16; gWl[i] += 16;
            // Speculative warm-up one further tile ahead (faults are dropped).
            __builtin_prefetch(gXh[i], 0, 1);
            __builtin_prefetch(gWh[i], 0, 1);
        }
    };

    v8f acc[2][4];
#pragma unroll
    for (int m = 0; m < 2; ++m)
#pragma unroll
        for (int n = 0; n < 4; ++n)
#pragma unroll
            for (int r = 0; r < 8; ++r) acc[m][n][r] = 0.0f;

    stage(0);                               // prologue: tile 0 -> buffer 0

    const int NK = DIN_ / BK;               // 64 K-steps
    for (int k = 0; k < NK; ++k) {
        const int cur = k & 1;
        const bool hasNext = (k + 1) < NK;
        if (hasNext) {
            stage((cur ^ 1) * TILEB);       // tile k+1 -> other buffer
            // <=8 outstanding: the 8 just issued; tile k's loads are done.
            asm volatile("s_wait_asynccnt 0x8" ::: "memory");
        } else {
            asm volatile("s_wait_asynccnt 0x0" ::: "memory");
        }
        __syncthreads();                    // buffer `cur` visible to all waves

        // ---- Fragments: pure packed-bf16 LDS reads from buffer `cur`. ----
        const unsigned bufu = cur * TILEU;
        v16bf ahi[2], alo[2], bhi[4], blo[4];
#pragma unroll
        for (int m = 0; m < 2; ++m) {
            const unsigned* rh = As_hi + bufu + (waveM * 32 + m * 16 + l16) * LDU;
            const unsigned* rl = As_lo + bufu + (waveM * 32 + m * 16 + l16) * LDU;
            v8u h, l;
#pragma unroll
            for (int j = 0; j < 4; ++j) {
                h[j]     = rh[4 * half + j];
                h[4 + j] = rh[8 + 4 * half + j];
                l[j]     = rl[4 * half + j];
                l[4 + j] = rl[8 + 4 * half + j];
            }
            ahi[m] = __builtin_bit_cast(v16bf, h);
            alo[m] = __builtin_bit_cast(v16bf, l);
        }
#pragma unroll
        for (int n = 0; n < 4; ++n) {
            const unsigned* rh = Bs_hi + bufu + (waveN * 64 + n * 16 + l16) * LDU;
            const unsigned* rl = Bs_lo + bufu + (waveN * 64 + n * 16 + l16) * LDU;
            v8u h, l;
#pragma unroll
            for (int j = 0; j < 4; ++j) {
                h[j]     = rh[4 * half + j];
                h[4 + j] = rh[8 + 4 * half + j];
                l[j]     = rl[4 * half + j];
                l[4 + j] = rl[8 + 4 * half + j];
            }
            bhi[n] = __builtin_bit_cast(v16bf, h);
            blo[n] = __builtin_bit_cast(v16bf, l);
        }

        // ---- 3-pass WMMA: hh + hl + lh. ----
#pragma unroll
        for (int m = 0; m < 2; ++m)
#pragma unroll
            for (int n = 0; n < 4; ++n) {
                acc[m][n] = __builtin_amdgcn_wmma_f32_16x16x32_bf16(
                    false, ahi[m], false, bhi[n], (short)0, acc[m][n], false, false);
                acc[m][n] = __builtin_amdgcn_wmma_f32_16x16x32_bf16(
                    false, ahi[m], false, blo[n], (short)0, acc[m][n], false, false);
                acc[m][n] = __builtin_amdgcn_wmma_f32_16x16x32_bf16(
                    false, alo[m], false, bhi[n], (short)0, acc[m][n], false, false);
            }
        __syncthreads();   // all waves done reading `cur` before k+1 overwrites it
    }

    // Epilogue: add bias, store f32 base directly into d_out.
#pragma unroll
    for (int n = 0; n < 4; ++n) {
        int col = blockN + waveN * 64 + n * 16 + l16;
        float bv = bias[col];
#pragma unroll
        for (int m = 0; m < 2; ++m) {
            int row0 = blockM + waveM * 32 + m * 16 + half * 8;  // C/D: VGPR r -> M = r + 8*half
#pragma unroll
            for (int r = 0; r < 8; ++r)
                out[(size_t)(row0 + r) * H_ + col] = acc[m][n][r] + bv;
        }
    }
}

// ---------------------------------------------------------------------------
// scale = max(|base[:, anchors, :E]|, 1e-6), one block.
// ---------------------------------------------------------------------------
__global__ void scale_kernel(const float* __restrict__ base, float* __restrict__ scale) {
    __shared__ float red[256];
    float m = 0.0f;
    for (int i = threadIdx.x; i < B_ * NA_ * E_; i += 256) {
        int bi  = i / (NA_ * E_);
        int rem = i % (NA_ * E_);
        int ai  = rem / E_;
        int e   = rem % E_;
        int t   = ai * 4 + 3;
        m = fmaxf(m, fabsf(base[((size_t)bi * T_ + t) * H_ + e]));
    }
    red[threadIdx.x] = m;
    __syncthreads();
    for (int s = 128; s > 0; s >>= 1) {
        if (threadIdx.x < s) red[threadIdx.x] = fmaxf(red[threadIdx.x], red[threadIdx.x + s]);
        __syncthreads();
    }
    if (threadIdx.x == 0) scale[0] = fmaxf(red[0], 1e-6f);
}

// ---------------------------------------------------------------------------
// Per-anchor: softmax over E, top-2 soft gate, w @ LiMEs -> p_mix[b,a,RANK].
// ---------------------------------------------------------------------------
__global__ void pmix_kernel(const float* __restrict__ base, const float* __restrict__ scale,
                            const float* __restrict__ limes, float* __restrict__ pmix) {
    int idx = blockIdx.x * blockDim.x + threadIdx.x;
    if (idx >= B_ * NA_) return;
    int bi = idx / NA_, ai = idx % NA_;
    int t = ai * 4 + 3;
    float inv = 1.0f / scale[0];   // TEMP == 1.0

    float l[E_];
    float mx = -1e30f;
#pragma unroll
    for (int e = 0; e < E_; ++e) {
        l[e] = base[((size_t)bi * T_ + t) * H_ + e] * inv;
        mx = fmaxf(mx, l[e]);
    }
    float sum = 0.0f;
#pragma unroll
    for (int e = 0; e < E_; ++e) { l[e] = expf(l[e] - mx); sum += l[e]; }
    float p[E_], m1 = -1.0f, m2 = -1.0f;
#pragma unroll
    for (int e = 0; e < E_; ++e) {
        p[e] = l[e] / sum;
        if (p[e] > m1) { m2 = m1; m1 = p[e]; } else if (p[e] > m2) { m2 = p[e]; }
    }
    float w[E_], wsum = 0.0f;
#pragma unroll
    for (int e = 0; e < E_; ++e) {
        float mask = 1.0f / (1.0f + expf(-(p[e] - m2) * 2.0f));  // 1/SOFT_T = 2
        w[e] = p[e] * mask;
        wsum += w[e];
    }
    float invw = 1.0f / (wsum + 1e-9f);
#pragma unroll
    for (int e = 0; e < E_; ++e) w[e] *= invw;

    for (int r = 0; r < RANK_; ++r) {
        float a = 0.0f;
#pragma unroll
        for (int e = 0; e < E_; ++e) a += w[e] * limes[e * RANK_ + r];
        pmix[(size_t)idx * RANK_ + r] = a;
    }
}

// ---------------------------------------------------------------------------
// out[..., :RANK] *= (1 + scaling * p_mix_full); rest already in place.
// ---------------------------------------------------------------------------
__global__ void apply_kernel(const float* __restrict__ pmix, float* __restrict__ out) {
    int i = blockIdx.x * blockDim.x + threadIdx.x;
    if (i >= B_ * T_ * RANK_) return;
    int bi  = i / (T_ * RANK_);
    int rem = i % (T_ * RANK_);
    int t   = rem / RANK_;
    int r   = rem % RANK_;
    int a   = t >> 2;   // NGRAM=4, T%4==0
    float pm = pmix[((size_t)bi * NA_ + a) * RANK_ + r];
    size_t off = ((size_t)bi * T_ + t) * H_ + r;
    out[off] = out[off] * (1.0f + SCALING_ * pm);
}

extern "C" void kernel_launch(void* const* d_in, const int* in_sizes, int n_in,
                              void* d_out, int out_size, void* d_ws, size_t ws_size,
                              hipStream_t stream) {
    (void)in_sizes; (void)n_in; (void)out_size; (void)ws_size;
    const float* x     = (const float*)d_in[0];
    const float* W     = (const float*)d_in[1];
    const float* b     = (const float*)d_in[2];
    const float* limes = (const float*)d_in[3];
    float* out   = (float*)d_out;
    float* scale = (float*)d_ws;           // 1 float
    float* pmix  = (float*)d_ws + 64;      // B*NA*RANK floats = 512 KB

    const size_t XBYTES = (size_t)B_ * T_ * DIN_ * 2;  // 32 MB per half
    const size_t WBYTES = (size_t)H_ * DIN_ * 2;       //  8 MB per half
    char* wsb = (char*)d_ws;
    unsigned* Xh = (unsigned*)(wsb + (1u << 20));
    unsigned* Xl = (unsigned*)(wsb + (1u << 20) + XBYTES);
    unsigned* Wh = (unsigned*)(wsb + (1u << 20) + 2 * XBYTES);
    unsigned* Wl = (unsigned*)(wsb + (1u << 20) + 2 * XBYTES + WBYTES);

    const int XN8 = (B_ * T_ * DIN_) / 8;  // 8 f32 per prep thread
    const int WN8 = (H_ * DIN_) / 8;
    split_kernel<<<(XN8 + 255) / 256, 256, 0, stream>>>(x, Xh, Xl, XN8);
    split_kernel<<<(WN8 + 255) / 256, 256, 0, stream>>>(W, Wh, Wl, WN8);

    dim3 grid(H_ / BN, (B_ * T_) / BM);    // 16 x 64 blocks
    gemm_kernel<<<grid, 256, 0, stream>>>(Xh, Xl, Wh, Wl, b, out);
    scale_kernel<<<1, 256, 0, stream>>>(out, scale);
    pmix_kernel<<<(B_ * NA_ + 255) / 256, 256, 0, stream>>>(out, scale, limes, pmix);
    apply_kernel<<<(B_ * T_ * RANK_ + 255) / 256, 256, 0, stream>>>(pmix, out);
}
